// MultimodalFusionLayer_7541962571977
// MI455X (gfx1250) — matrix operand order repaired and checked by reading, not compiled, not run on hardware
//
#include <hip/hip_runtime.h>

typedef __attribute__((ext_vector_type(8)))  float   v8f;
typedef __attribute__((ext_vector_type(2)))  float   v2f;
typedef __attribute__((ext_vector_type(16))) __bf16  v16bf;

#define Bsz    128
#define WCOLS  232      // 4 left pad + 224 + 4 right pad
#define LROWS  21       // input rows staged per workgroup (16 + 7 - 2 halo)

__device__ __forceinline__ unsigned short f2bf(float f) {
    unsigned int u = __float_as_uint(f);
    u += 0x7fffu + ((u >> 16) & 1u);           // round-to-nearest-even
    return (unsigned short)(u >> 16);
}

// ---------------------------------------------------------------------------
// Kernel 0: pre-swizzle conv weights into CDNA5 bf16 B-fragment layout.
// K reindexed as K' = (c*7+ky)*8 + p where p=0 is a zero-pad column and
// p=1..7 map to kx=0..6; padded to 192 = 6 chunks of 32.  This pad-at-front
// choice makes the activation LDS left pad 4, so staging stores are 8B
// aligned while A-fragment gathers stay 4B aligned (start pix = 2*ox).
// B frag layout (32x16 bf16): lane<16 -> N=lane, elements e=0..15 hold
// K = kc*32+e ; lane>=16 -> N=lane-16, K = kc*32+16+e.
// ---------------------------------------------------------------------------
__global__ void __launch_bounds__(256) build_bpack(
        const float* __restrict__ conv_w, unsigned short* __restrict__ bpack) {
    int i = blockIdx.x * 256 + threadIdx.x;
    if (i >= 6 * 4 * 32 * 16) return;
    int e  = i & 15;
    int l  = (i >> 4) & 31;
    int nt = (i >> 9) & 3;
    int kc = i >> 11;
    int n  = nt * 16 + (l & 15);
    int kp = kc * 32 + (l >> 4) * 16 + e;      // K' index
    int g  = kp >> 3;                          // group = c*7 + ky
    int p  = kp & 7;                           // p=0 pad, p-1 = kx
    float v = 0.f;
    if (g < 21 && p >= 1) {
        int c = g / 7, ky = g - c * 7;
        v = conv_w[((n * 3 + c) * 7 + ky) * 7 + (p - 1)];   // OIHW
    }
    bpack[i] = f2bf(v);
}

// ---------------------------------------------------------------------------
// Kernel 1: fused Conv(3->64,k7,s2,p3) + bias + ReLU + MaxPool2x2 + spatial sum
// via implicit-GEMM bf16 WMMA.  One workgroup = (image b, block of 4 pooled
// rows = 8 conv rows).  4 waves, each wave processes 7 (row-pair, col-tile)
// units of 16 conv columns; per unit: 2 rows x 6 K-chunks x 4 N-tiles WMMA.
// Writes partial spatial sums [B][14][64]; no atomics (deterministic).
// ---------------------------------------------------------------------------
__global__ void __launch_bounds__(128) conv_fused(
        const float* __restrict__ img,              // [128,3,224,224]
        const unsigned short* __restrict__ bpack,   // [6][4][32][16] bf16
        const float* __restrict__ conv_b,           // [64]
        float* __restrict__ partial)                // [128][14][64]
{
    __shared__ __align__(16) unsigned short in_lds[3 * LROWS * WCOLS]; // 29232 B
    __shared__ __align__(16) unsigned short bq_lds[6 * 4 * 32 * 16];   // 24576 B
    __shared__ float wsum[4 * 4 * 32];                                 //  2048 B

    const int rblock = blockIdx.x;      // 0..13
    const int b      = blockIdx.y;      // 0..127
    const int tid    = threadIdx.x;
    const int lane   = tid & 31;
    const int wave   = tid >> 5;
    const int half   = lane >> 4;       // 0/1 lane half
    const int mcol   = lane & 15;

    // ---- phase 1: zero padded input region (b128 stores) ------------------
    {
        uint4* z = (uint4*)in_lds;
        const uint4 z4 = {0u, 0u, 0u, 0u};
        for (int i = tid; i < (3 * LROWS * WCOLS) / 8; i += 128) z[i] = z4;
    }
    __syncthreads();

    // ---- phase 2: stage inputs (float4 -> 4x bf16 -> b64 store) -----------
    const int iy0 = 16 * rblock - 3;
    for (int i = tid; i < 3 * LROWS * 56; i += 128) {
        int c   = i / (LROWS * 56);
        int rem = i - c * LROWS * 56;
        int j   = rem / 56;
        int ix  = (rem - j * 56) * 4;
        int iy  = iy0 + j;
        if (iy >= 0 && iy < 224) {
            const float4 v = *(const float4*)&img[((b * 3 + c) * 224 + iy) * 224 + ix];
            uint2 pk;
            pk.x = (unsigned int)f2bf(v.x) | ((unsigned int)f2bf(v.y) << 16);
            pk.y = (unsigned int)f2bf(v.z) | ((unsigned int)f2bf(v.w) << 16);
            *(uint2*)&in_lds[(c * LROWS + j) * WCOLS + ix + 4] = pk;   // 8B aligned
        }
    }
    {
        const uint4* bsrc = (const uint4*)bpack;
        uint4*       bdst = (uint4*)bq_lds;
        for (int i = tid; i < 1536; i += 128) bdst[i] = bsrc[i];
    }
    __syncthreads();

    // per-lane channel bias (channel N = nt*16 + mcol is constant per lane)
    float bias[4];
#pragma unroll
    for (int nt = 0; nt < 4; ++nt) bias[nt] = conv_b[nt * 16 + mcol];

    float sums[4] = {0.f, 0.f, 0.f, 0.f};
    const v8f vzero = {0.f, 0.f, 0.f, 0.f, 0.f, 0.f, 0.f, 0.f};

    for (int u = wave; u < 28; u += 4) {
        const int rp = u / 7;           // row-pair 0..3 -> conv rows oy0,oy0+1
        const int ct = u - rp * 7;      // column tile 0..6 (16 conv cols each)

        v8f acc[2][4];
#pragma unroll
        for (int r = 0; r < 2; ++r)
#pragma unroll
            for (int nt = 0; nt < 4; ++nt) acc[r][nt] = vzero;

#pragma unroll
        for (int kc = 0; kc < 6; ++kc) {
            // A fragments (16x32 bf16): elements 0..7  -> K'=kc*32+half*16+e,
            //                            elements 8..15 -> K'=kc*32+half*16+8+e
            // each 8-run = one (c,ky) group, contiguous in LDS from pix=2*ox.
            v16bf afrag[2];
            int g1 = kc * 4 + 2 * half;      if (g1 > 20) g1 = 20;  // B has zeros there
            int g2 = kc * 4 + 2 * half + 1;  if (g2 > 20) g2 = 20;
            int c1 = g1 / 7, ky1 = g1 - c1 * 7;
            int c2 = g2 / 7, ky2 = g2 - c2 * 7;
#pragma unroll
            for (int r = 0; r < 2; ++r) {
                union { unsigned int u[8]; v16bf v; } au;
                const unsigned int* p1 = (const unsigned int*)
                    &in_lds[(c1 * LROWS + (4 * rp + 2 * r + ky1)) * WCOLS + 32 * ct + 2 * mcol];
                const unsigned int* p2 = (const unsigned int*)
                    &in_lds[(c2 * LROWS + (4 * rp + 2 * r + ky2)) * WCOLS + 32 * ct + 2 * mcol];
                au.u[0] = p1[0]; au.u[1] = p1[1]; au.u[2] = p1[2]; au.u[3] = p1[3];
                au.u[4] = p2[0]; au.u[5] = p2[1]; au.u[6] = p2[2]; au.u[7] = p2[3];
                afrag[r] = au.v;
            }
#pragma unroll
            for (int nt = 0; nt < 4; ++nt) {
                union { unsigned int u[8]; v16bf v; } bu;
                const unsigned int* bp = (const unsigned int*)
                    &bq_lds[((kc * 4 + nt) * 32 + lane) * 16];
                bu.u[0] = bp[0]; bu.u[1] = bp[1]; bu.u[2] = bp[2]; bu.u[3] = bp[3];
                bu.u[4] = bp[4]; bu.u[5] = bp[5]; bu.u[6] = bp[6]; bu.u[7] = bp[7];
                acc[0][nt] = __builtin_amdgcn_wmma_f32_16x16x32_bf16(
                    false, afrag[0], false, bu.v, (short)0, acc[0][nt], false, false);
                acc[1][nt] = __builtin_amdgcn_wmma_f32_16x16x32_bf16(
                    false, afrag[1], false, bu.v, (short)0, acc[1][nt], false, false);
            }
        }

        // bias + ReLU + 2x2 maxpool (rows = acc[0]/acc[1], col pairs = element
        // pairs since M = elem + 8*half), then accumulate spatial sum.
#pragma unroll
        for (int nt = 0; nt < 4; ++nt) {
            float rowm[8];
#pragma unroll
            for (int j = 0; j < 8; ++j) {
                float a0 = acc[0][nt][j] + bias[nt]; a0 = a0 > 0.f ? a0 : 0.f;
                float a1 = acc[1][nt][j] + bias[nt]; a1 = a1 > 0.f ? a1 : 0.f;
                rowm[j] = fmaxf(a0, a1);
            }
#pragma unroll
            for (int t = 0; t < 4; ++t)
                sums[nt] += fmaxf(rowm[2 * t], rowm[2 * t + 1]);
        }
    }

    // deterministic block reduction (no float atomics)
#pragma unroll
    for (int nt = 0; nt < 4; ++nt) wsum[(wave * 4 + nt) * 32 + lane] = sums[nt];
    __syncthreads();
    if (tid < 64) {
        int nt = tid >> 4, q = tid & 15;
        float s = 0.f;
#pragma unroll
        for (int w = 0; w < 4; ++w) {
            s += wsum[(w * 4 + nt) * 32 + q];
            s += wsum[(w * 4 + nt) * 32 + q + 16];
        }
        partial[(b * 14 + rblock) * 64 + tid] = s;
    }
}

// Kernel 2: finish adaptive-avg-pool: pooled[b][c] = sum(partials)/3136
__global__ void __launch_bounds__(256) pool_finish(
        const float* __restrict__ partial, float* __restrict__ pooled) {
    int i = blockIdx.x * 256 + threadIdx.x;
    if (i >= Bsz * 64) return;
    int b = i >> 6, c = i & 63;
    float s = 0.f;
    for (int r = 0; r < 14; ++r) s += partial[(b * 14 + r) * 64 + c];
    pooled[i] = s * (1.0f / 3136.0f);
}

// ---------------------------------------------------------------------------
// Generic fp32 WMMA GEMM:  Y[M,N] = X[M,K] @ W[N,K]^T + bias (+ addend)
// One wave per 16x16 tile; K stepped by 4 via v_wmma_f32_16x16x4_f32.
// ---------------------------------------------------------------------------
__global__ void __launch_bounds__(32) gemm_xwt_f32(
        const float* __restrict__ X, const float* __restrict__ W,
        const float* __restrict__ bias, const float* __restrict__ addend,
        float* __restrict__ Y, int M, int N, int K)
{
    const int nt   = blockIdx.x, mt = blockIdx.y;
    const int lane = threadIdx.x & 31;
    const int half = lane >> 4, q = lane & 15;
    const int mrow = mt * 16 + q;
    const int col  = nt * 16 + q;
    const bool colok = col < N;
    const int wrow = colok ? col : 0;

    v8f acc = {0.f, 0.f, 0.f, 0.f, 0.f, 0.f, 0.f, 0.f};
    for (int k0 = 0; k0 < K; k0 += 4) {
        v2f a, bb;
        a.x = X[mrow * K + k0 + half * 2];
        a.y = X[mrow * K + k0 + half * 2 + 1];
        float w0 = W[wrow * K + k0 + half * 2];
        float w1 = W[wrow * K + k0 + half * 2 + 1];
        bb.x = colok ? w0 : 0.f;
        bb.y = colok ? w1 : 0.f;
        acc = __builtin_amdgcn_wmma_f32_16x16x4_f32(
            false, a, false, bb, (short)0, acc, false, false);
    }
    if (colok) {
        float bv = bias ? bias[col] : 0.f;
#pragma unroll
        for (int j = 0; j < 8; ++j) {
            int row = mt * 16 + j + 8 * half;
            float v = acc[j] + bv;
            if (addend) v += addend[row * N + col];
            Y[row * N + col] = v;
        }
    }
}

// LayerNorm over rows of 512: one wave per row.
__global__ void __launch_bounds__(32) ln_rows(
        const float* __restrict__ h, const float* __restrict__ g,
        const float* __restrict__ bt, float* __restrict__ out) {
    int row = blockIdx.x, lane = threadIdx.x;
    float s = 0.f, ss = 0.f;
    for (int i = lane; i < 512; i += 32) {
        float v = h[row * 512 + i];
        s += v; ss += v * v;
    }
#pragma unroll
    for (int off = 16; off; off >>= 1) {
        s  += __shfl_xor(s,  off, 32);
        ss += __shfl_xor(ss, off, 32);
    }
    float mu   = s * (1.0f / 512.0f);
    float var  = ss * (1.0f / 512.0f) - mu * mu;
    float rstd = rsqrtf(var + 1e-5f);
    for (int i = lane; i < 512; i += 32) {
        float v = (h[row * 512 + i] - mu) * rstd;
        out[row * 512 + i] = v * g[i] + bt[i];
    }
}

// concat [fused(512), lang(512), gnn(32), joint(4)] -> [128,1060]
__global__ void __launch_bounds__(256) concat_feats(
        const float* __restrict__ fused, const float* __restrict__ lang,
        const float* __restrict__ gnn,   const float* __restrict__ joint,
        float* __restrict__ cc) {
    int i = blockIdx.x * 256 + threadIdx.x;
    if (i >= Bsz * 1060) return;
    int row = i / 1060, col = i - row * 1060;
    float v;
    if      (col < 512)  v = fused[row * 512 + col];
    else if (col < 1024) v = lang [row * 512 + col - 512];
    else if (col < 1056) v = gnn  [row * 32  + col - 1024];
    else                 v = joint[row * 4   + col - 1056];
    cc[i] = v;
}

extern "C" void kernel_launch(void* const* d_in, const int* in_sizes, int n_in,
                              void* d_out, int out_size, void* d_ws, size_t ws_size,
                              hipStream_t stream) {
    const float* vision  = (const float*)d_in[0];
    const float* lang    = (const float*)d_in[1];
    const float* gnn     = (const float*)d_in[2];
    const float* joint   = (const float*)d_in[3];
    const float* conv_w  = (const float*)d_in[4];
    const float* conv_b  = (const float*)d_in[5];
    const float* vis_w   = (const float*)d_in[6];
    const float* vis_b   = (const float*)d_in[7];
    const float* graph_w = (const float*)d_in[8];
    const float* graph_b = (const float*)d_in[9];
    // d_in[10..13] = wq,bq,wk,bk: softmax over a single key == 1 -> attn == V,
    // so the Q/K projections are algebraically dead. Skip them.
    const float* wv      = (const float*)d_in[14];
    const float* bv      = (const float*)d_in[15];
    const float* wo      = (const float*)d_in[16];
    const float* bo      = (const float*)d_in[17];
    const float* ln_g    = (const float*)d_in[18];
    const float* ln_b    = (const float*)d_in[19];
    const float* fin_w   = (const float*)d_in[20];
    const float* fin_b   = (const float*)d_in[21];
    float* out = (float*)d_out;

    char* ws = (char*)d_ws;
    size_t off = 0;
    auto alloc = [&](size_t bytes) -> void* {
        void* p = ws + off;
        off = (off + bytes + 255) & ~(size_t)255;
        return p;
    };
    unsigned short* bpack   = (unsigned short*)alloc(6 * 4 * 32 * 16 * sizeof(unsigned short));
    float* partial    = (float*)alloc((size_t)Bsz * 14 * 64 * sizeof(float));
    float* pooled     = (float*)alloc((size_t)Bsz * 64 * sizeof(float));
    float* vis_feat   = (float*)alloc((size_t)Bsz * 512 * sizeof(float));
    float* graph_feat = (float*)alloc((size_t)Bsz * 512 * sizeof(float));
    float* vbuf       = (float*)alloc((size_t)Bsz * 512 * sizeof(float));
    float* hbuf       = (float*)alloc((size_t)Bsz * 512 * sizeof(float));
    float* fused      = (float*)alloc((size_t)Bsz * 512 * sizeof(float));
    float* cc         = (float*)alloc((size_t)Bsz * 1060 * sizeof(float));

    // 1) swizzle conv weights into bf16 B fragments
    build_bpack<<<48, 256, 0, stream>>>(conv_w, bpack);
    // 2) fused conv+relu+maxpool+avgpool partials (the 30 GFLOP hot spot)
    conv_fused<<<dim3(14, Bsz), 128, 0, stream>>>(vision, bpack, conv_b, partial);
    pool_finish<<<(Bsz * 64 + 255) / 256, 256, 0, stream>>>(partial, pooled);
    // 3) vis_feat = pooled @ vis_w^T + vis_b           [128,512]
    gemm_xwt_f32<<<dim3(32, 8), 32, 0, stream>>>(pooled, vis_w, vis_b, nullptr,
                                                 vis_feat, Bsz, 512, 64);
    // 4) graph_feat = gnn @ graph_w^T + graph_b        [128,512]
    gemm_xwt_f32<<<dim3(32, 8), 32, 0, stream>>>(gnn, graph_w, graph_b, nullptr,
                                                 graph_feat, Bsz, 512, 32);
    // 5) v = graph_feat @ wv^T + bv                    [128,512]
    gemm_xwt_f32<<<dim3(32, 8), 32, 0, stream>>>(graph_feat, wv, bv, nullptr,
                                                 vbuf, Bsz, 512, 512);
    // 6) h = vis_feat + (v @ wo^T + bo)                [128,512]
    gemm_xwt_f32<<<dim3(32, 8), 32, 0, stream>>>(vbuf, wo, bo, vis_feat,
                                                 hbuf, Bsz, 512, 512);
    // 7) fused = LayerNorm(h) * ln_g + ln_b
    ln_rows<<<Bsz, 32, 0, stream>>>(hbuf, ln_g, ln_b, fused);
    // 8) concat + final projection                      [128,548]
    concat_feats<<<(Bsz * 1060 + 255) / 256, 256, 0, stream>>>(fused, lang, gnn, joint, cc);
    gemm_xwt_f32<<<dim3(35, 8), 32, 0, stream>>>(cc, fin_w, fin_b, nullptr,
                                                 out, Bsz, 548, 1060);
}
